// SelfAttention_5437428596840
// MI455X (gfx1250) — matrix operand order repaired and checked by reading, not compiled, not run on hardware
//
#include <hip/hip_runtime.h>

typedef __attribute__((ext_vector_type(16))) _Float16 v16h;
typedef __attribute__((ext_vector_type(8)))  _Float16 v8h;
typedef __attribute__((ext_vector_type(8)))  float    v8f;

constexpr int kB  = 16;    // batch
constexpr int kC  = 512;   // channels
constexpr int kN  = 1024;  // H*W
constexpr int kC3 = 1536;  // 3*C
constexpr int kG  = 32;    // groups
constexpr int kCPG = 16;   // channels per group

// GEMM tiling: 256 threads = 8 waves; block tile 128(M) x 128(N), K step 32.
// Each wave: one 16-row M strip x 128 N cols = 8 WMMA accumulators.
constexpr int BM = 128;
constexpr int BN = 128;
constexpr int BK = 32;
constexpr int NG = BN / 16;     // 8 n-groups per wave
constexpr int LDS_STRIDE = 40;  // halves: 32 data + 8 pad (80B rows, 16B aligned)

__device__ __forceinline__ v16h cat8(v8h lo, v8h hi) {
  return __builtin_shufflevector(lo, hi, 0, 1, 2, 3, 4, 5, 6, 7,
                                 8, 9, 10, 11, 12, 13, 14, 15);
}

// LDS byte address: generic LDS pointer = {shared aperture, 32-bit LDS offset}
__device__ __forceinline__ uint32_t lds_addr32(const void* p) {
  return (uint32_t)(size_t)p;
}

// Async DMA: global -> LDS, 16 bytes per lane, tracked by ASYNCcnt.
__device__ __forceinline__ void async_copy_b128(uint32_t lds, const void* gptr) {
#if defined(__gfx1250__)
  asm volatile("global_load_async_to_lds_b128 %0, %1, off"
               :: "v"(lds), "v"(gptr)
               : "memory");
#endif
}

__device__ __forceinline__ void wait_async0() {
#if defined(__gfx1250__)
  asm volatile("s_wait_asynccnt 0x0" ::: "memory");
#endif
}

// Core: acc[NG] (each 16x16 f32) for one wave strip of a 128x128 block tile.
// A: row-major [M][K] f16 (lda). Bt: B transposed, row-major [N][K] f16 (ldb).
// Double-buffered LDS, async global->LDS DMA pipelined against WMMA.
__device__ __forceinline__ void wmma_tile_gemm(
    const _Float16* __restrict__ A, int lda,
    const _Float16* __restrict__ Bt, int ldb,
    int K, int tileM, int tileN, v8f acc[NG]) {
  __shared__ __align__(16) _Float16 As[2][BM * LDS_STRIDE];
  __shared__ __align__(16) _Float16 Bs[2][BN * LDS_STRIDE];

  const int tid  = threadIdx.x;
  const int lane = tid & 31;
  const int wave = tid >> 5;

#pragma unroll
  for (int g = 0; g < NG; ++g) acc[g] = v8f{};

  // staging: uniform across all 256 threads (A & B: 32B/thread each)
  const int arow = tid >> 1;               // 0..127
  const int aseg = (tid & 1) * 16;         // half offset 0 or 16

  const _Float16* agp = A + (size_t)(tileM + arow) * lda + aseg;
  const _Float16* bgp = Bt + (size_t)(tileN + arow) * ldb + aseg;

  const uint32_t lA0 = lds_addr32(&As[0][arow * LDS_STRIDE + aseg]);
  const uint32_t lA1 = lds_addr32(&As[1][arow * LDS_STRIDE + aseg]);
  const uint32_t lB0 = lds_addr32(&Bs[0][arow * LDS_STRIDE + aseg]);
  const uint32_t lB1 = lds_addr32(&Bs[1][arow * LDS_STRIDE + aseg]);

  // WMMA fragment addressing (ISA 7.12.2 layouts)
  const int m_lane = wave * 16 + (lane & 15);   // A row this lane owns
  const int a_kb   = (lane >> 4) * 8;           // A: K sub-block base (halves)
  const int b_col  = lane & 15;                 // B column within 16-col group
  const int b_kh   = (lane >> 4) * 8;           // B: K sub-block base (halves)

  // prologue: DMA tile 0 into buffer 0
  async_copy_b128(lA0, agp);
  async_copy_b128(lA0 + 32, agp + 16);  // +16 halves = +32 bytes
  async_copy_b128(lB0, bgp);
  async_copy_b128(lB0 + 32, bgp + 16);

  int buf = 0;
  for (int k0 = 0; k0 < K; k0 += BK, buf ^= 1) {
    wait_async0();      // this wave's DMA into buffer `buf` has landed
    __syncthreads();    // every wave's DMA landed; other buffer free to fill

    if (k0 + BK < K) {  // DMA next tile into the other buffer, overlapped
      const _Float16* ag = agp + k0 + BK;
      const _Float16* bg = bgp + k0 + BK;
      const uint32_t nA = buf ? lA0 : lA1;
      const uint32_t nB = buf ? lB0 : lB1;
      async_copy_b128(nA, ag);
      async_copy_b128(nA + 32, ag + 16);
      async_copy_b128(nB, bg);
      async_copy_b128(nB + 32, bg + 16);
    }

    const _Float16* asb = As[buf];
    const _Float16* bsb = Bs[buf];
    const v8h alo = *(const v8h*)&asb[m_lane * LDS_STRIDE + a_kb];
    const v8h ahi = *(const v8h*)&asb[m_lane * LDS_STRIDE + 16 + a_kb];
    const v16h af = cat8(alo, ahi);
    v16h bf[NG];
#pragma unroll
    for (int g = 0; g < NG; ++g) {
      const v8h blo = *(const v8h*)&bsb[(g * 16 + b_col) * LDS_STRIDE + b_kh];
      const v8h bhi = *(const v8h*)&bsb[(g * 16 + b_col) * LDS_STRIDE + 16 + b_kh];
      bf[g] = cat8(blo, bhi);
    }
#pragma unroll
    for (int g = 0; g < NG; ++g) {
      acc[g] = __builtin_amdgcn_wmma_f32_16x16x32_f16(
          false, af, false, bf[g], (short)0, acc[g], false, false);
    }
    // scheduler: batch all DS reads, then the WMMA chain (one dscnt wait)
    __builtin_amdgcn_sched_group_barrier(0x100, 18, 0);  // DS reads
    __builtin_amdgcn_sched_group_barrier(0x008, NG, 0);  // WMMA
  }
}

// ---------------- stage kernels ----------------

__global__ void __launch_bounds__(256)
f32_to_f16_kernel(const float* __restrict__ src, _Float16* __restrict__ dst,
                  int n) {
  const int i = blockIdx.x * 256 + threadIdx.x;
  if (i < n) dst[i] = (_Float16)src[i];
}

// GroupNorm: one block per (b, group); writes xn f16 as [b][n][c].
__global__ void __launch_bounds__(256)
groupnorm_kernel(const float* __restrict__ x, const float* __restrict__ gamma,
                 const float* __restrict__ beta, _Float16* __restrict__ xn16) {
  const int bg = blockIdx.x;
  const int b = bg >> 5;
  const int g = bg & 31;
  const float* xp = x + ((size_t)b * kC + g * kCPG) * kN;
  const int tid = threadIdx.x;

  float s = 0.f, q = 0.f;
  for (int i = tid; i < kCPG * kN; i += 256) {
    const float v = xp[i];
    s += v;
    q += v * v;
  }
  __shared__ float rs[256], rq[256];
  rs[tid] = s;
  rq[tid] = q;
  __syncthreads();
  for (int t = 128; t > 0; t >>= 1) {
    if (tid < t) {
      rs[tid] += rs[tid + t];
      rq[tid] += rq[tid + t];
    }
    __syncthreads();
  }
  const float invn = 1.0f / (float)(kCPG * kN);
  const float mean = rs[0] * invn;
  const float var  = rq[0] * invn - mean * mean;
  const float rstd = rsqrtf(var + 1e-5f);

  for (int i = tid; i < kCPG * kN; i += 256) {
    const int cl = i >> 10;          // i / kN
    const int n  = i & (kN - 1);     // i % kN
    const int c  = g * kCPG + cl;
    const float v = (xp[i] - mean) * rstd * gamma[c] + beta[c];
    xn16[((size_t)b * kN + n) * kC + c] = (_Float16)v;
  }
}

// QKV GEMM: qkv[o][n] = w_qkv[o][:] . xn[:][n]; split-store q/k [b][n][c], v [b][c][n].
__global__ void __launch_bounds__(256)
qkv_gemm_kernel(const _Float16* __restrict__ w16,
                const _Float16* __restrict__ xn16,
                const float* __restrict__ b_qkv, _Float16* __restrict__ q16,
                _Float16* __restrict__ k16, _Float16* __restrict__ v16) {
  const int b = blockIdx.z;
  const int tileM = blockIdx.y * BM;
  const int tileN = blockIdx.x * BN;
  v8f acc[NG];
  wmma_tile_gemm(w16, kC, xn16 + (size_t)b * kN * kC, kC, kC, tileM, tileN, acc);

  const int lane = threadIdx.x & 31;
  const int wave = threadIdx.x >> 5;
  const int mrow0 = tileM + wave * 16 + ((lane >> 4) << 3);
  const int ncol0 = tileN + (lane & 15);
#pragma unroll
  for (int g = 0; g < NG; ++g) {
    const int n = ncol0 + g * 16;
#pragma unroll
    for (int r = 0; r < 8; ++r) {
      const int o = mrow0 + r;
      const float val = acc[g][r] + b_qkv[o];
      if (o < kC) {
        q16[((size_t)b * kN + n) * kC + o] = (_Float16)val;
      } else if (o < 2 * kC) {
        k16[((size_t)b * kN + n) * kC + (o - kC)] = (_Float16)val;
      } else {
        v16[((size_t)b * kC + (o - 2 * kC)) * kN + n] = (_Float16)val;
      }
    }
  }
}

// Scores: S[i][j] = scale * q[i][:] . k[j][:]  (both stored [n][c])
__global__ void __launch_bounds__(256)
scores_gemm_kernel(const _Float16* __restrict__ q16,
                   const _Float16* __restrict__ k16, float* __restrict__ S,
                   float scale) {
  const int b = blockIdx.z;
  const int tileM = blockIdx.y * BM;  // i
  const int tileN = blockIdx.x * BN;  // j
  v8f acc[NG];
  wmma_tile_gemm(q16 + (size_t)b * kN * kC, kC, k16 + (size_t)b * kN * kC, kC,
                 kC, tileM, tileN, acc);

  const int lane = threadIdx.x & 31;
  const int wave = threadIdx.x >> 5;
  const int i0 = tileM + wave * 16 + ((lane >> 4) << 3);
  const int j0 = tileN + (lane & 15);
#pragma unroll
  for (int g = 0; g < NG; ++g) {
    const int j = j0 + g * 16;
#pragma unroll
    for (int r = 0; r < 8; ++r) {
      S[((size_t)b * kN + (i0 + r)) * kN + j] = acc[g][r] * scale;
    }
  }
}

// Row softmax over 1024 f32, emits f16 attn [b][i][j].
__global__ void __launch_bounds__(256)
softmax_kernel(const float* __restrict__ S, _Float16* __restrict__ attn16) {
  const size_t row = blockIdx.x;  // b*kN + i
  const float* sp = S + row * kN;
  const int tid = threadIdx.x;
  __shared__ float red[256];

  const float v0 = sp[tid];
  const float v1 = sp[tid + 256];
  const float v2 = sp[tid + 512];
  const float v3 = sp[tid + 768];
  red[tid] = fmaxf(fmaxf(v0, v1), fmaxf(v2, v3));
  __syncthreads();
  for (int t = 128; t > 0; t >>= 1) {
    if (tid < t) red[tid] = fmaxf(red[tid], red[tid + t]);
    __syncthreads();
  }
  const float m = red[0];
  __syncthreads();

  const float e0 = __expf(v0 - m);
  const float e1 = __expf(v1 - m);
  const float e2 = __expf(v2 - m);
  const float e3 = __expf(v3 - m);
  red[tid] = e0 + e1 + e2 + e3;
  __syncthreads();
  for (int t = 128; t > 0; t >>= 1) {
    if (tid < t) red[tid] += red[tid + t];
    __syncthreads();
  }
  const float inv = 1.0f / red[0];
  _Float16* op = attn16 + row * kN;
  op[tid]       = (_Float16)(e0 * inv);
  op[tid + 256] = (_Float16)(e1 * inv);
  op[tid + 512] = (_Float16)(e2 * inv);
  op[tid + 768] = (_Float16)(e3 * inv);
}

// A*V: out[c][i] = v[c][:] . attn[i][:]; store [b][i][c] f16.
__global__ void __launch_bounds__(256)
av_gemm_kernel(const _Float16* __restrict__ v16,
               const _Float16* __restrict__ attn16,
               _Float16* __restrict__ out16) {
  const int b = blockIdx.z;
  const int tileM = blockIdx.y * BM;  // c
  const int tileN = blockIdx.x * BN;  // i
  v8f acc[NG];
  wmma_tile_gemm(v16 + (size_t)b * kC * kN, kN, attn16 + (size_t)b * kN * kN,
                 kN, kN, tileM, tileN, acc);

  const int lane = threadIdx.x & 31;
  const int wave = threadIdx.x >> 5;
  const int c0 = tileM + wave * 16 + ((lane >> 4) << 3);
  const int i0 = tileN + (lane & 15);
#pragma unroll
  for (int g = 0; g < NG; ++g) {
    const int i = i0 + g * 16;
#pragma unroll
    for (int r = 0; r < 8; ++r) {
      out16[((size_t)b * kN + i) * kC + (c0 + r)] = (_Float16)acc[g][r];
    }
  }
}

// Projection + residual: y[b][o][n] = x[b][o][n] + w_proj[o][:] . out[:][n] + b_proj[o]
__global__ void __launch_bounds__(256)
proj_gemm_kernel(const _Float16* __restrict__ wp16,
                 const _Float16* __restrict__ out16,
                 const float* __restrict__ b_proj, const float* __restrict__ x,
                 float* __restrict__ y) {
  const int b = blockIdx.z;
  const int tileM = blockIdx.y * BM;  // o
  const int tileN = blockIdx.x * BN;  // n
  v8f acc[NG];
  wmma_tile_gemm(wp16, kC, out16 + (size_t)b * kN * kC, kC, kC, tileM, tileN,
                 acc);

  const int lane = threadIdx.x & 31;
  const int wave = threadIdx.x >> 5;
  const int o0 = tileM + wave * 16 + ((lane >> 4) << 3);
  const int n0 = tileN + (lane & 15);
#pragma unroll
  for (int g = 0; g < NG; ++g) {
    const int n = n0 + g * 16;
#pragma unroll
    for (int r = 0; r < 8; ++r) {
      const int o = o0 + r;
      const size_t idx = ((size_t)b * kC + o) * kN + n;
      y[idx] = x[idx] + acc[g][r] + b_proj[o];
    }
  }
}

// ---------------- launch ----------------

extern "C" void kernel_launch(void* const* d_in, const int* in_sizes, int n_in,
                              void* d_out, int out_size, void* d_ws,
                              size_t ws_size, hipStream_t stream) {
  const float* x      = (const float*)d_in[0];
  const float* gamma  = (const float*)d_in[1];
  const float* beta   = (const float*)d_in[2];
  const float* w_qkv  = (const float*)d_in[3];
  const float* b_qkv  = (const float*)d_in[4];
  const float* w_proj = (const float*)d_in[5];
  const float* b_proj = (const float*)d_in[6];
  float* out = (float*)d_out;

  char* ws = (char*)d_ws;
  size_t off = 0;
  auto alloc = [&](size_t bytes) -> void* {
    void* p = ws + off;
    off = (off + bytes + 255) & ~(size_t)255;
    return p;
  };

  _Float16* wqkv16  = (_Float16*)alloc((size_t)kC3 * kC * 2);
  _Float16* wproj16 = (_Float16*)alloc((size_t)kC * kC * 2);
  _Float16* xn16    = (_Float16*)alloc((size_t)kB * kN * kC * 2);
  _Float16* q16     = (_Float16*)alloc((size_t)kB * kN * kC * 2);
  _Float16* k16     = (_Float16*)alloc((size_t)kB * kN * kC * 2);
  _Float16* v16     = (_Float16*)alloc((size_t)kB * kC * kN * 2);
  float*    S       = (float*)   alloc((size_t)kB * kN * kN * 4);
  _Float16* attn16  = (_Float16*)alloc((size_t)kB * kN * kN * 2);
  _Float16* out16   = (_Float16*)alloc((size_t)kB * kN * kC * 2);

  // 1. weights -> f16
  {
    const int n1 = kC3 * kC;
    f32_to_f16_kernel<<<(n1 + 255) / 256, 256, 0, stream>>>(w_qkv, wqkv16, n1);
    const int n2 = kC * kC;
    f32_to_f16_kernel<<<(n2 + 255) / 256, 256, 0, stream>>>(w_proj, wproj16, n2);
  }
  // 2. GroupNorm -> xn16 [b][n][c]
  groupnorm_kernel<<<kB * kG, 256, 0, stream>>>(x, gamma, beta, xn16);
  // 3. QKV GEMM
  qkv_gemm_kernel<<<dim3(kN / BN, kC3 / BM, kB), 256, 0, stream>>>(
      wqkv16, xn16, b_qkv, q16, k16, v16);
  // 4. scores = scale * q k^T
  const float scale = 1.0f / sqrtf((float)kC);
  scores_gemm_kernel<<<dim3(kN / BN, kN / BM, kB), 256, 0, stream>>>(
      q16, k16, S, scale);
  // 5. softmax rows
  softmax_kernel<<<kB * kN, 256, 0, stream>>>(S, attn16);
  // 6. out = v attn^T
  av_gemm_kernel<<<dim3(kN / BN, kC / BM, kB), 256, 0, stream>>>(
      v16, attn16, out16);
  // 7. projection + residual
  proj_gemm_kernel<<<dim3(kN / BN, kC / BM, kB), 256, 0, stream>>>(
      wproj16, out16, b_proj, x, out);
}
